// DeformConv_50878182588423
// MI455X (gfx1250) — compile-verified, matrix operand order checked by reading, not run on hardware
//
#include <hip/hip_runtime.h>
#include <math.h>

// ---------- CDNA5 WMMA types ----------
typedef __attribute__((ext_vector_type(16))) __bf16        v16bf;
typedef __attribute__((ext_vector_type(8)))  float         v8f;
typedef __attribute__((ext_vector_type(8)))  unsigned int  v8u;

#define HW      4096
#define KSTEPS  72      // K = 9*256 = 2304, 72 steps of 32

// round-to-nearest-even fp32 -> bf16, packed pair
__device__ __forceinline__ unsigned int pack2bf(float a, float b) {
    unsigned int ua = __builtin_bit_cast(unsigned int, a);
    unsigned int ub = __builtin_bit_cast(unsigned int, b);
    ua = (ua + 0x7FFFu + ((ua >> 16) & 1u)) >> 16;
    ub = (ub + 0x7FFFu + ((ub >> 16) & 1u)) >> 16;
    return (ua & 0xFFFFu) | (ub << 16);
}

__device__ __forceinline__ v8f wmma_bf16(v8u a, v8u b, v8f c) {
    return __builtin_amdgcn_wmma_f32_16x16x32_bf16(
        false, __builtin_bit_cast(v16bf, a),
        false, __builtin_bit_cast(v16bf, b),
        (short)0, c, false, false);
}

// ============================================================
// Kernel P: pack weights into B-fragment swizzled bf16 layout,
//           fold conv bias + batchnorm into scale/shift.
// K index ordering: kidx = ktap*256 + ci  (ci fastest within a K-step)
// B frag layout (ISA 7.12.2, 16-bit): lane = n + 16*h, vgpr i holds
// bf16 pair at k0 = (i&3)*2 + h*8 + (i>>2)*16 within the 32-K step.
// ============================================================
__global__ __launch_bounds__(256) void pack_kernel(
    const float* __restrict__ w, const float* __restrict__ w_off,
    const float* __restrict__ bias, const float* __restrict__ gamma,
    const float* __restrict__ beta, const float* __restrict__ mean,
    const float* __restrict__ var,
    unsigned int* __restrict__ wbp_main, unsigned int* __restrict__ wbp_off,
    float* __restrict__ fscale, float* __restrict__ fshift)
{
    const int NMAIN = KSTEPS * 16 * 32 * 8;  // 294912
    const int NOFF  = KSTEPS *  2 * 32 * 8;  //  36864
    int tid = blockIdx.x * 256 + threadIdx.x;

    if (tid < NMAIN) {
        int i    = tid & 7;
        int lane = (tid >> 3) & 31;
        int nt   = (tid >> 8) & 15;
        int kt   = tid >> 12;
        int h    = lane >> 4;
        int n    = nt * 16 + (lane & 15);
        int k0   = kt * 32 + (i & 3) * 2 + h * 8 + (i >> 2) * 16;
        // w layout: [Co][Ci][3][3];  B[kidx][n] = w[n][ci][ktap]
        float a  = w[(n * 256 + (k0 & 255)) * 9 + (k0 >> 8)];
        float b2 = w[(n * 256 + ((k0 + 1) & 255)) * 9 + ((k0 + 1) >> 8)];
        wbp_main[tid] = pack2bf(a, b2);
    } else if (tid < NMAIN + NOFF) {
        int t    = tid - NMAIN;
        int i    = t & 7;
        int lane = (t >> 3) & 31;
        int nj   = (t >> 8) & 1;
        int kt   = t >> 9;
        int h    = lane >> 4;
        int n    = nj * 16 + (lane & 15);
        int k0   = kt * 32 + (i & 3) * 2 + h * 8 + (i >> 2) * 16;
        float a = 0.f, b2 = 0.f;
        if (n < 27) {   // pad 27 offset channels to 32 with zeros
            a  = w_off[(n * 256 + (k0 & 255)) * 9 + (k0 >> 8)];
            b2 = w_off[(n * 256 + ((k0 + 1) & 255)) * 9 + ((k0 + 1) >> 8)];
        }
        wbp_off[t] = pack2bf(a, b2);
    } else if (tid < NMAIN + NOFF + 256) {
        int co = tid - NMAIN - NOFF;
        float sc = gamma[co] * rsqrtf(var[co] + 1e-5f);
        fscale[co] = sc;
        fshift[co] = beta[co] + (bias[co] - mean[co]) * sc;
    }
}

// ============================================================
// Kernel O: offset conv as WMMA GEMM.  M=32768 pixels, N=32 (27 pad),
// K=2304 (im2col of x, zero padding).  WG: 256 thr = 8 waves,
// tile M=64 x N=32: wave (mi in 0..3, nj in 0..1).
// ============================================================
__global__ __launch_bounds__(256) void offset_gemm(
    const float* __restrict__ x, const unsigned int* __restrict__ wbp_off,
    const float* __restrict__ b_off, float* __restrict__ om)
{
    __shared__ unsigned int As[64 * 16];     // [pixel][k-pair] bf16x2
    const int t      = threadIdx.x;
    const int Mbase  = blockIdx.x * 64;      // pixels, tile never crosses batch
    const int b      = Mbase >> 12;
    const int lane   = t & 31;
    const int wv     = t >> 5;
    const int mi     = wv & 3;
    const int nj     = wv >> 2;
    const int h      = lane >> 4;
    const int ln     = lane & 15;

    // A-tile build mapping: 64 pixels x 32 ci = 2048 elems, 8 per thread
    const int p  = t >> 2;                   // pixel in tile (0..63)
    const int cg = (t & 3) * 8;              // 8 consecutive ci
    const int yx = (Mbase & 4095) + p;
    const int py = yx >> 6, px = yx & 63;

    v8f acc = {};

    for (int kt = 0; kt < KSTEPS; ++kt) {
        const int ktap = kt >> 3;
        const int cb   = (kt & 7) * 32;
        const int oy   = py + (ktap / 3) - 1;
        const int ox   = px + (ktap % 3) - 1;
        const bool valid = (oy >= 0) && (oy < 64) && (ox >= 0) && (ox < 64);
        const int sidx = valid ? (oy * 64 + ox) : 0;
        const float vs = valid ? 1.f : 0.f;
        const float* xb = x + (((size_t)((b << 8) + cb + cg)) << 12) + sidx;
        float v[8];
        #pragma unroll
        for (int j = 0; j < 8; ++j) v[j] = xb[(size_t)j << 12] * vs;
        #pragma unroll
        for (int j = 0; j < 4; ++j)
            As[p * 16 + (cg >> 1) + j] = pack2bf(v[2 * j], v[2 * j + 1]);
        __syncthreads();

        v8u af = {};
        const int arow = mi * 16 + ln;
        #pragma unroll
        for (int i = 0; i < 8; ++i)
            af[i] = As[arow * 16 + (i & 3) + h * 4 + (i >> 2) * 8];

        const unsigned int* bp = wbp_off + ((size_t)((kt * 2 + nj) * 32 + lane) << 3);
        v8u bfv = {};
        #pragma unroll
        for (int i = 0; i < 8; ++i) bfv[i] = bp[i];

        acc = wmma_bf16(af, bfv, acc);
        __syncthreads();
    }

    // epilogue: C layout -> om[b][ch][yx], add offset bias
    const int ch = nj * 16 + ln;
    const float bo = (ch < 27) ? b_off[ch] : 0.f;
    #pragma unroll
    for (int v2 = 0; v2 < 8; ++v2) {
        const int m = Mbase + mi * 16 + h * 8 + v2;
        om[((size_t)((b << 5) + ch)) * HW + (m & 4095)] = acc[v2] + bo;
    }
}

// ============================================================
// Kernel M: per (b, ktap, pixel) -> bilinear corner indices and
// mask-premultiplied weights (validity folded in).  32B records.
// ============================================================
__global__ __launch_bounds__(256) void meta_kernel(
    const float* __restrict__ om, unsigned int* __restrict__ meta)
{
    const int t    = blockIdx.x * 256 + threadIdx.x;  // < 8*9*4096
    const int yx   = t & 4095;
    const int bk   = t >> 12;
    const int ktap = bk % 9;
    const int b    = bk / 9;

    const float dy = om[((size_t)((b << 5) +      ktap)) * HW + yx];
    const float dx = om[((size_t)((b << 5) +  9 + ktap)) * HW + yx];
    const float ml = om[((size_t)((b << 5) + 18 + ktap)) * HW + yx];
    const float msk = 1.f / (1.f + __expf(-ml));

    const int ypix = yx >> 6, xpix = yx & 63;
    const float py = (float)ypix + (float)(ktap / 3 - 1) + dy;
    const float px = (float)xpix + (float)(ktap % 3 - 1) + dx;
    const float y0f = floorf(py), x0f = floorf(px);
    const float wy = py - y0f, wx = px - x0f;
    const int y0 = (int)y0f, x0 = (int)x0f;

    const float cw[4] = {(1.f - wy) * (1.f - wx), (1.f - wy) * wx,
                         wy * (1.f - wx),          wy * wx};
    const int yy[4] = {y0, y0, y0 + 1, y0 + 1};
    const int xx[4] = {x0, x0 + 1, x0, x0 + 1};

    unsigned int idx[4]; float wt[4];
    #pragma unroll
    for (int c = 0; c < 4; ++c) {
        const bool vld = (yy[c] >= 0) && (yy[c] < 64) && (xx[c] >= 0) && (xx[c] < 64);
        const int yc = min(max(yy[c], 0), 63);
        const int xc = min(max(xx[c], 0), 63);
        idx[c] = (unsigned)(yc * 64 + xc);
        wt[c]  = vld ? cw[c] * msk : 0.f;
    }

    unsigned int* mp = meta + ((size_t)t << 3);
    uint4  iv = make_uint4(idx[0], idx[1], idx[2], idx[3]);
    float4 wv4 = make_float4(wt[0], wt[1], wt[2], wt[3]);
    *(uint4*)mp        = iv;
    *(float4*)(mp + 4) = wv4;
}

// ============================================================
// Kernel G: main WMMA GEMM with on-the-fly bilinear gather.
// M=32768, N=256, K=2304.  WG: 256 thr = 8 waves,
// tile M=64 x N=256.  Wave (mg in 0..1, ng in 0..3) register-blocks
// 2 M-tiles x 4 N-tiles -> 8 accumulators, 8 WMMAs per K-step;
// each B fragment load feeds 2 WMMAs (halves global B traffic).
// ============================================================
__global__ __launch_bounds__(256) void main_gemm(
    const float* __restrict__ x,
    const unsigned int* __restrict__ wbp,
    const unsigned int* __restrict__ meta,
    const float* __restrict__ fscale, const float* __restrict__ fshift,
    float* __restrict__ out)
{
    __shared__ unsigned int As[64 * 16];     // [pixel][k-pair]
    const int t     = threadIdx.x;
    const int Mbase = blockIdx.x * 64;
    const int b     = Mbase >> 12;
    const int lane  = t & 31;
    const int wv    = t >> 5;
    const int ng    = wv & 3;                // N group: 4 N-tiles (64 cols)
    const int mg    = wv >> 2;               // M group: 2 M-tiles (32 rows)
    const int h     = lane >> 4;
    const int ln    = lane & 15;

    // A-tile build: 64 pixels x 32 ci = 2048 elems, 8 per thread
    const int p  = t >> 2;                   // pixel in tile (0..63)
    const int cg = (t & 3) * 8;              // 8 consecutive ci
    const int yx = (Mbase & 4095) + p;

    v8f acc[2][4] = {};

    for (int kt = 0; kt < KSTEPS; ++kt) {
        const int ktap = kt >> 3;
        const int cb   = (kt & 7) * 32;

        // ---- build A tile (bilinear gather, mask folded into weights) ----
        const unsigned int* mp = meta + ((((size_t)(b * 9 + ktap) << 12) + yx) << 3);
        const uint4  mi4 = *(const uint4*)(mp);
        const float4 mw  = *(const float4*)(mp + 4);

        const float* xb = x + (((size_t)((b << 8) + cb + cg)) << 12);
        float vv[8];
        #pragma unroll
        for (int j = 0; j < 8; ++j) {
            const float* xc = xb + ((size_t)j << 12);
            vv[j] = mw.x * xc[mi4.x] + mw.y * xc[mi4.y]
                  + mw.z * xc[mi4.z] + mw.w * xc[mi4.w];
        }
        #pragma unroll
        for (int j = 0; j < 4; ++j)
            As[p * 16 + (cg >> 1) + j] = pack2bf(vv[2 * j], vv[2 * j + 1]);
        __syncthreads();

        // ---- A fragments for this wave's two M-tiles ----
        v8u af[2];
        #pragma unroll
        for (int mi2 = 0; mi2 < 2; ++mi2) {
            const int arow = mg * 32 + mi2 * 16 + ln;
            #pragma unroll
            for (int i = 0; i < 8; ++i)
                af[mi2][i] = As[arow * 16 + (i & 3) + h * 4 + (i >> 2) * 8];
        }

        // ---- 4 B fragments, each reused across both M-tiles ----
        #pragma unroll
        for (int j = 0; j < 4; ++j) {
            const int nt = ng * 4 + j;
            const unsigned int* bp = wbp + ((size_t)((kt * 16 + nt) * 32 + lane) << 3);
            v8u bfv = {};
            #pragma unroll
            for (int i = 0; i < 8; ++i) bfv[i] = bp[i];
            acc[0][j] = wmma_bf16(af[0], bfv, acc[0][j]);
            acc[1][j] = wmma_bf16(af[1], bfv, acc[1][j]);
        }
        __syncthreads();
    }

    // epilogue: fused (conv bias + BN) scale/shift + ReLU
    #pragma unroll
    for (int j = 0; j < 4; ++j) {
        const int co = (ng * 4 + j) * 16 + ln;
        const float sc = fscale[co];
        const float sh = fshift[co];
        #pragma unroll
        for (int mi2 = 0; mi2 < 2; ++mi2) {
            #pragma unroll
            for (int v2 = 0; v2 < 8; ++v2) {
                const int m = Mbase + mg * 32 + mi2 * 16 + h * 8 + v2;
                const float r = acc[mi2][j][v2] * sc + sh;
                out[(((size_t)((b << 8) + co)) << 12) + (m & 4095)] = fmaxf(r, 0.f);
            }
        }
    }
}

// ============================================================
extern "C" void kernel_launch(void* const* d_in, const int* in_sizes, int n_in,
                              void* d_out, int out_size, void* d_ws, size_t ws_size,
                              hipStream_t stream)
{
    const float* x     = (const float*)d_in[0];
    const float* w_off = (const float*)d_in[1];
    const float* b_off = (const float*)d_in[2];
    const float* w     = (const float*)d_in[3];
    const float* bias  = (const float*)d_in[4];
    const float* gamma = (const float*)d_in[5];
    const float* beta  = (const float*)d_in[6];
    const float* mean  = (const float*)d_in[7];
    const float* var   = (const float*)d_in[8];
    float* out = (float*)d_out;

    char* ws = (char*)d_ws;
    unsigned int* wbp_main = (unsigned int*)ws;  ws += (size_t)KSTEPS * 16 * 32 * 8 * 4; // 1.18 MB
    unsigned int* wbp_off  = (unsigned int*)ws;  ws += (size_t)KSTEPS *  2 * 32 * 8 * 4; // 147 KB
    float* fscale          = (float*)ws;         ws += 256 * 4;
    float* fshift          = (float*)ws;         ws += 256 * 4;
    float* om              = (float*)ws;         ws += (size_t)8 * 32 * HW * 4;          // 4.19 MB
    unsigned int* meta     = (unsigned int*)ws;  ws += (size_t)8 * 9 * HW * 32;          // 9.44 MB

    pack_kernel<<<1297, 256, 0, stream>>>(w, w_off, bias, gamma, beta, mean, var,
                                          wbp_main, wbp_off, fscale, fshift);
    offset_gemm<<<512, 256, 0, stream>>>(x, wbp_off, b_off, om);
    meta_kernel<<<1152, 256, 0, stream>>>(om, meta);
    main_gemm<<<512, 256, 0, stream>>>(x, wbp_main, meta, fscale, fshift, out);
}
